// SSM_49632642072963
// MI455X (gfx1250) — matrix operand order repaired and checked by reading, not compiled, not run on hardware
//
#include <hip/hip_runtime.h>
#include <hip/hip_bf16.h>
#include <stdint.h>

// ---------------------------------------------------------------------------
// Problem constants (from reference setup_inputs)
// ---------------------------------------------------------------------------
#define BB   2
#define LL   4096
#define DD   2048
#define NN   2048
#define ROWS (BB * LL)        // 8192
#define NCH  64               // scan chunks per sequence
#define CHL  (LL / NCH)       // 64 steps per chunk
#define EPSV 1e-5f

// ---------------------------------------------------------------------------
// WMMA types (gfx1250 / CDNA5)
// ---------------------------------------------------------------------------
typedef __attribute__((ext_vector_type(16))) __bf16 v16bf;
typedef __attribute__((ext_vector_type(8)))  float  v8f;

union FragU {
    v16bf v;
    uint4 q[2];
};

// f32 -> bf16 with round-to-nearest-even (stored as raw u16)
__device__ __forceinline__ uint16_t f2bf(float f) {
    uint32_t u = __float_as_uint(f);
    u += 0x7FFFu + ((u >> 16) & 1u);
    return (uint16_t)(u >> 16);
}

__device__ __forceinline__ float sigmoidf_dev(float x) {
    return 1.0f / (1.0f + __expf(-x));
}

// A fragment: 16x32 (MxK) bf16 tile from row-major [*, ld] matrix.
// lane L holds row M = m0 + (L&15); K = k0 + ((L>=16)?8:0) + {0..7, 16..23}
__device__ __forceinline__ v16bf load_fragA(const uint16_t* base, int ld,
                                            int m0, int k0, int lane) {
    const int row  = m0 + (lane & 15);
    const int koff = (lane >> 4) * 8;
    const uint16_t* p = base + (size_t)row * ld + k0 + koff;
    FragU f;
    f.q[0] = *reinterpret_cast<const uint4*>(p);        // K = kb .. kb+7
    f.q[1] = *reinterpret_cast<const uint4*>(p + 16);   // K = kb+16 .. kb+23
    return f.v;
}

// B fragment: 32x16 (KxN) where B[k][n] = W[n][k], W row-major [*, ld].
// lane L holds col N = n0 + (L&15); K = k0 + ((L>=16)?8:0) + {0..7, 16..23}
__device__ __forceinline__ v16bf load_fragB(const uint16_t* base, int ld,
                                            int n0, int k0, int lane) {
    const int row  = n0 + (lane & 15);
    const int koff = (lane >> 4) * 8;
    const uint16_t* p = base + (size_t)row * ld + k0 + koff;
    FragU f;
    f.q[0] = *reinterpret_cast<const uint4*>(p);        // K = kb .. kb+7
    f.q[1] = *reinterpret_cast<const uint4*>(p + 16);   // K = kb+16 .. kb+23
    return f.v;
}

// ---------------------------------------------------------------------------
// Kernel 0: fp32 -> bf16 weight conversion
// ---------------------------------------------------------------------------
__global__ void k_f32_to_bf16(const float* __restrict__ src,
                              uint16_t* __restrict__ dst, int count) {
    int i = blockIdx.x * blockDim.x + threadIdx.x;
    if (i < count) dst[i] = f2bf(src[i]);
}

// ---------------------------------------------------------------------------
// Kernel 1: RMSNorm + bf16 cast.  One block per row (D = 2048, 256 threads).
// ---------------------------------------------------------------------------
__global__ void k_rmsnorm(const float* __restrict__ x,
                          const float* __restrict__ rms_w,
                          uint16_t* __restrict__ xn) {
    __shared__ float red[256];
    const int row = blockIdx.x;
    const float* xr = x + (size_t)row * DD;
    float ss = 0.0f;
    #pragma unroll
    for (int i = 0; i < DD / 256; ++i) {
        float v = xr[threadIdx.x + i * 256];
        ss += v * v;
    }
    red[threadIdx.x] = ss;
    __syncthreads();
    for (int s = 128; s > 0; s >>= 1) {
        if (threadIdx.x < s) red[threadIdx.x] += red[threadIdx.x + s];
        __syncthreads();
    }
    const float inv = rsqrtf(red[0] / (float)DD + EPSV);
    uint16_t* outr = xn + (size_t)row * DD;
    #pragma unroll
    for (int i = 0; i < DD / 256; ++i) {
        int d = threadIdx.x + i * 256;
        outr[d] = f2bf(xr[d] * rms_w[d] * inv);
    }
}

// ---------------------------------------------------------------------------
// Kernel 2: fused 5-way GEMM (A,B,Ca,Cb,D projections) + gates.
//   xn[ROWS][DD] (bf16) x W[5][NN][DD] (bf16) -> a,u,c fp32 [ROWS][NN]
//   Wave tile: 64(M) x 16(N); per K-step: 4 A-frags + 5 B-frags feed 20 WMMAs.
//   Block 128 threads = 4 waves covering 64(m) x 64(n).
// ---------------------------------------------------------------------------
__global__ void __launch_bounds__(128, 1)
k_gemm5_gates(const uint16_t* __restrict__ xn,
              const uint16_t* __restrict__ w5,   // [5][NN][DD]
              const float* __restrict__ A_b,
              float* __restrict__ a_out,
              float* __restrict__ u_out,
              float* __restrict__ c_out) {
    const int lane = threadIdx.x & 31;
    const int wave = threadIdx.x >> 5;
    const int m0 = blockIdx.x * 64;                 // 4 m-subtiles per wave
    const int n0 = blockIdx.y * 64 + wave * 16;

    v8f acc[4][5];
    #pragma unroll
    for (int mi = 0; mi < 4; ++mi)
        #pragma unroll
        for (int mat = 0; mat < 5; ++mat)
            acc[mi][mat] = v8f{};

    for (int k0 = 0; k0 < DD; k0 += 32) {
        if (k0 + 32 < DD) {
            // warm next A panel via global_prefetch_b8
            __builtin_prefetch(xn + (size_t)(m0 + (lane & 15)) * DD + k0 + 32, 0, 1);
        }
        v16bf fa[4];
        #pragma unroll
        for (int mi = 0; mi < 4; ++mi)
            fa[mi] = load_fragA(xn, DD, m0 + mi * 16, k0, lane);

        #pragma unroll
        for (int mat = 0; mat < 5; ++mat) {
            v16bf fb = load_fragB(w5 + (size_t)mat * NN * DD, DD, n0, k0, lane);
            #pragma unroll
            for (int mi = 0; mi < 4; ++mi)
                acc[mi][mat] = __builtin_amdgcn_wmma_f32_16x16x32_bf16(
                    false, fa[mi], false, fb, (short)0, acc[mi][mat], false, false);
        }
    }

    // Epilogue: C/D layout -> lane holds col n0+(lane&15), rows m+j+8*(lane>=16)
    const int nCol  = n0 + (lane & 15);
    const float bias = A_b[nCol];
    #pragma unroll
    for (int mi = 0; mi < 4; ++mi) {
        const int mBase = m0 + mi * 16 + ((lane >> 4) * 8);
        #pragma unroll
        for (int j = 0; j < 8; ++j) {
            const size_t idx = (size_t)(mBase + j) * NN + nCol;
            const float va  = sigmoidf_dev(acc[mi][0][j] + bias);
            const float vb  = acc[mi][1][j];
            const float vca = acc[mi][2][j];
            const float vcb = acc[mi][3][j];
            const float vd  = sigmoidf_dev(acc[mi][4][j]);
            a_out[idx] = va;
            u_out[idx] = vd * vb;                          // input-gated scan input
            c_out[idx] = (vca * sigmoidf_dev(vca)) * vcb;  // silu(ca) * cb
        }
    }
}

// ---------------------------------------------------------------------------
// Kernel 3a: per-chunk scan reduction. thread = (b, chunk, n).
//   (Aprod, Ucomb) over CHL steps.
// ---------------------------------------------------------------------------
__global__ void k_scan_chunk(const float* __restrict__ a,
                             const float* __restrict__ u,
                             float* __restrict__ carryA,
                             float* __restrict__ carryU) {
    const int tid = blockIdx.x * blockDim.x + threadIdx.x;
    const int n  = tid % NN;
    const int ch = (tid / NN) % NCH;
    const int b  = tid / (NN * NCH);
    const size_t base = ((size_t)(b * LL + ch * CHL)) * NN + n;
    float Ap = 1.0f, U = 0.0f;
    for (int l = 0; l < CHL; ++l) {
        const size_t idx = base + (size_t)l * NN;
        const float av = a[idx];
        U  = fmaf(av, U, u[idx]);
        Ap *= av;
    }
    const size_t cidx = ((size_t)b * NCH + ch) * NN + n;
    carryA[cidx] = Ap;
    carryU[cidx] = U;
}

// ---------------------------------------------------------------------------
// Kernel 3b: serial scan over chunk carries. thread = (b, n).
//   prefixH[b][ch][n] = exclusive h entering chunk ch.
// ---------------------------------------------------------------------------
__global__ void k_scan_carries(const float* __restrict__ carryA,
                               const float* __restrict__ carryU,
                               float* __restrict__ prefixH) {
    const int tid = blockIdx.x * blockDim.x + threadIdx.x;
    const int n = tid % NN;
    const int b = tid / NN;
    float h = 0.0f;
    for (int ch = 0; ch < NCH; ++ch) {
        const size_t idx = ((size_t)b * NCH + ch) * NN + n;
        prefixH[idx] = h;
        h = fmaf(carryA[idx], h, carryU[idx]);
    }
}

// ---------------------------------------------------------------------------
// Kernel 3c: replay chunk with prefix, emit y = c * h as bf16.
// ---------------------------------------------------------------------------
__global__ void k_scan_apply(const float* __restrict__ a,
                             const float* __restrict__ u,
                             const float* __restrict__ c,
                             const float* __restrict__ prefixH,
                             uint16_t* __restrict__ y) {
    const int tid = blockIdx.x * blockDim.x + threadIdx.x;
    const int n  = tid % NN;
    const int ch = (tid / NN) % NCH;
    const int b  = tid / (NN * NCH);
    const size_t base = ((size_t)(b * LL + ch * CHL)) * NN + n;
    float h = prefixH[((size_t)b * NCH + ch) * NN + n];
    for (int l = 0; l < CHL; ++l) {
        const size_t idx = base + (size_t)l * NN;
        h = fmaf(a[idx], h, u[idx]);
        y[idx] = f2bf(c[idx] * h);
    }
}

// ---------------------------------------------------------------------------
// Kernel 4: output GEMM  out = y @ out_w^T + out_b + x   (WMMA bf16)
//   y[ROWS][NN] bf16, out_w_bf[DD][NN], out fp32 [ROWS][DD]
//   Wave tile: 64(M) x 16(N); 4 WMMAs per B-frag.
// ---------------------------------------------------------------------------
__global__ void __launch_bounds__(128, 1)
k_gemm_out(const uint16_t* __restrict__ y,
           const uint16_t* __restrict__ wOut,
           const float* __restrict__ out_b,
           const float* __restrict__ x,
           float* __restrict__ out) {
    const int lane = threadIdx.x & 31;
    const int wave = threadIdx.x >> 5;
    const int m0 = blockIdx.x * 64;
    const int d0 = blockIdx.y * 64 + wave * 16;

    v8f acc[4];
    #pragma unroll
    for (int mi = 0; mi < 4; ++mi) acc[mi] = v8f{};

    for (int k0 = 0; k0 < NN; k0 += 32) {
        if (k0 + 32 < NN) {
            __builtin_prefetch(y + (size_t)(m0 + (lane & 15)) * NN + k0 + 32, 0, 1);
        }
        v16bf fb = load_fragB(wOut, NN, d0, k0, lane);
        #pragma unroll
        for (int mi = 0; mi < 4; ++mi) {
            v16bf fa = load_fragA(y, NN, m0 + mi * 16, k0, lane);
            acc[mi] = __builtin_amdgcn_wmma_f32_16x16x32_bf16(
                false, fa, false, fb, (short)0, acc[mi], false, false);
        }
    }

    const int dCol  = d0 + (lane & 15);
    const float bias = out_b[dCol];
    #pragma unroll
    for (int mi = 0; mi < 4; ++mi) {
        const int mBase = m0 + mi * 16 + ((lane >> 4) * 8);
        #pragma unroll
        for (int j = 0; j < 8; ++j) {
            const size_t idx = (size_t)(mBase + j) * DD + dCol;
            out[idx] = acc[mi][j] + bias + x[idx];   // residual
        }
    }
}

// ---------------------------------------------------------------------------
// Host-side launch
// ---------------------------------------------------------------------------
extern "C" void kernel_launch(void* const* d_in, const int* in_sizes, int n_in,
                              void* d_out, int out_size, void* d_ws, size_t ws_size,
                              hipStream_t stream) {
    (void)in_sizes; (void)n_in; (void)out_size; (void)ws_size;

    const float* x     = (const float*)d_in[0];
    const float* rms_w = (const float*)d_in[1];
    const float* A_w   = (const float*)d_in[2];
    const float* A_b   = (const float*)d_in[3];
    const float* B_w   = (const float*)d_in[4];
    const float* Ca_w  = (const float*)d_in[5];
    const float* Cb_w  = (const float*)d_in[6];
    const float* D_w   = (const float*)d_in[7];
    const float* out_w = (const float*)d_in[8];
    const float* out_b = (const float*)d_in[9];
    float* out = (float*)d_out;

    // ---- workspace layout (bytes) ----
    char* ws = (char*)d_ws;
    size_t off = 0;
    auto take = [&](size_t bytes) {
        void* p = ws + off;
        off += (bytes + 255) & ~(size_t)255;
        return p;
    };
    uint16_t* xn      = (uint16_t*)take((size_t)ROWS * DD * 2);       // 32 MB
    uint16_t* w5      = (uint16_t*)take((size_t)5 * NN * DD * 2);     // 40 MB
    uint16_t* wOut    = (uint16_t*)take((size_t)DD * NN * 2);         //  8 MB
    float*    a_arr   = (float*)   take((size_t)ROWS * NN * 4);       // 64 MB
    float*    u_arr   = (float*)   take((size_t)ROWS * NN * 4);       // 64 MB
    float*    c_arr   = (float*)   take((size_t)ROWS * NN * 4);       // 64 MB
    float*    carryA  = (float*)   take((size_t)BB * NCH * NN * 4);   //  1 MB
    float*    carryU  = (float*)   take((size_t)BB * NCH * NN * 4);   //  1 MB
    float*    prefixH = (float*)   take((size_t)BB * NCH * NN * 4);   //  1 MB
    uint16_t* y_bf    = (uint16_t*)take((size_t)ROWS * NN * 2);       // 32 MB

    const int WN = NN * DD;                       // elements per weight matrix

    // 0) weight conversion fp32 -> bf16 (packed [A,B,Ca,Cb,D] + out)
    {
        const int bs = 256, gs = (WN + bs - 1) / bs;
        k_f32_to_bf16<<<gs, bs, 0, stream>>>(A_w,  w5 + 0 * (size_t)WN, WN);
        k_f32_to_bf16<<<gs, bs, 0, stream>>>(B_w,  w5 + 1 * (size_t)WN, WN);
        k_f32_to_bf16<<<gs, bs, 0, stream>>>(Ca_w, w5 + 2 * (size_t)WN, WN);
        k_f32_to_bf16<<<gs, bs, 0, stream>>>(Cb_w, w5 + 3 * (size_t)WN, WN);
        k_f32_to_bf16<<<gs, bs, 0, stream>>>(D_w,  w5 + 4 * (size_t)WN, WN);
        k_f32_to_bf16<<<gs, bs, 0, stream>>>(out_w, wOut, WN);
    }

    // 1) RMSNorm + bf16 cast
    k_rmsnorm<<<ROWS, 256, 0, stream>>>(x, rms_w, xn);

    // 2) fused 5-way WMMA GEMM + gates (64x64 block tile, 4 waves)
    {
        dim3 grid(ROWS / 64, NN / 64);
        k_gemm5_gates<<<grid, 128, 0, stream>>>(xn, w5, A_b, a_arr, u_arr, c_arr);
    }

    // 3) chunked linear-recurrence scan
    {
        const int total = BB * NCH * NN;          // 262144
        k_scan_chunk<<<total / 256, 256, 0, stream>>>(a_arr, u_arr, carryA, carryU);
        k_scan_carries<<<(BB * NN) / 256, 256, 0, stream>>>(carryA, carryU, prefixH);
        k_scan_apply<<<total / 256, 256, 0, stream>>>(a_arr, u_arr, c_arr, prefixH, y_bf);
    }

    // 4) output WMMA GEMM + bias + residual (64x64 block tile)
    {
        dim3 grid(ROWS / 64, DD / 64);
        k_gemm_out<<<grid, 128, 0, stream>>>(y_bf, wOut, out_b, x, out);
    }
}
